// CustomLSTMModel_83554293776390
// MI455X (gfx1250) — matrix operand order repaired and checked by reading, not compiled, not run on hardware
//
#include <hip/hip_runtime.h>
#include <hip/hip_bf16.h>

typedef __attribute__((ext_vector_type(16))) _Float16 v16h;
typedef __attribute__((ext_vector_type(8)))  _Float16 v8h;
typedef __attribute__((ext_vector_type(8)))  float    v8f;
typedef __attribute__((ext_vector_type(4)))  float    v4f;

union AFrag { v16h v; v8h h[2]; };

#define WMMA16(A, B, C) \
  __builtin_amdgcn_wmma_f32_16x16x32_f16(false, (A), false, (B), (short)0, (C), false, false)

__device__ __forceinline__ float sigm(float x) { return 1.0f / (1.0f + __expf(-x)); }
__device__ __forceinline__ float tanh_fast(float x) {
  float t = __expf(-2.0f * fabsf(x));
  float r = (1.0f - t) / (1.0f + t);
  return copysignf(r, x);
}

#define TT_  64   // total timesteps
#define TC_  16   // time chunk
#define NC_  4    // chunks
#define H_   32   // hidden

// One wave (32 threads) owns 16 sequences.
// LDS: hbuf  [TC][16 seq][32 H]  f16  (current chunk h values, overwritten layer by layer)
//      hcarry[3 layers][16][32]  f16  (h state carried across chunks per layer)
//      xbuf  [16 seq][64 t]      f32  (staged input tile)
__global__ __launch_bounds__(32) void lstm3_wmma_kernel(
    const float* __restrict__ x,
    const float* __restrict__ wih0, const float* __restrict__ whh0,
    const float* __restrict__ bih0, const float* __restrict__ bhh0,
    const float* __restrict__ wih1, const float* __restrict__ whh1,
    const float* __restrict__ bih1, const float* __restrict__ bhh1,
    const float* __restrict__ wih2, const float* __restrict__ whh2,
    const float* __restrict__ bih2, const float* __restrict__ bhh2,
    const float* __restrict__ fcw, const float* __restrict__ fcb,
    float* __restrict__ out)
{
  extern __shared__ char smem[];
  _Float16* hbuf   = (_Float16*)smem;                      // 16 KB
  _Float16* hcarry = (_Float16*)(smem + TC_ * 16 * H_ * 2);      // 3 KB
  float*    xbuf   = (float*)(smem + (TC_ + 3) * 16 * H_ * 2);   // 4 KB

  const int  lane = threadIdx.x;      // 0..31 (wave32)
  const int  half = lane >> 4;        // A/B fragment K-half selector
  const int  lmod = lane & 15;        // row (A) / column (B,C) index
  const long seq_base = (long)blockIdx.x * 16;

  const float* WIH[3] = {wih0, wih1, wih2};
  const float* WHH[3] = {whh0, whh1, whh2};
  const float* BIH[3] = {bih0, bih1, bih2};
  const float* BHH[3] = {bhh0, bhh1, bhh2};

  // ---- stage x tile: 16 seq x 64 t contiguous f32 (coalesced b128 loads) ----
  {
    const float* xg = x + seq_base * TT_;
    #pragma unroll
    for (int i = 0; i < 8; ++i) {
      int idx = (lane + i * 32) * 4;
      *(v4f*)(xbuf + idx) = *(const v4f*)(xg + idx);
    }
  }
  __syncthreads();

  // per-layer c state, C-matrix layout: element v ~ seq (v + 8*half), lane%16 ~ H (and +16)
  float c0s[3][8], c1s[3][8];
  #pragma unroll
  for (int l = 0; l < 3; ++l)
    #pragma unroll
    for (int v = 0; v < 8; ++v) { c0s[l][v] = 0.f; c1s[l][v] = 0.f; }

  for (int c = 0; c < NC_; ++c) {
    #pragma unroll
    for (int l = 0; l < 3; ++l) {
      // ---- build B fragments for this layer (W^T: K contiguous in row-major W) ----
      v16h Bhh[8];
      v16h Bih[8];
      float bias[8];
      float wx0[8];
      #pragma unroll
      for (int j = 0; j < 8; ++j) {
        const int n = j * 16 + lmod;              // gate column
        const float* wr = WHH[l] + n * H_ + half * 8;
        v16h b;
        #pragma unroll
        for (int e = 0; e < 16; ++e) b[e] = (_Float16)wr[e < 8 ? e : e + 8];
        Bhh[j] = b;
        bias[j] = BIH[l][n] + BHH[l][n];
        if (l == 0) {
          wx0[j] = WIH[0][n];                     // input dim 1: rank-1 contribution
        } else {
          const float* wi = WIH[l] + n * H_ + half * 8;
          v16h bi;
          #pragma unroll
          for (int e = 0; e < 16; ++e) bi[e] = (_Float16)wi[e < 8 ? e : e + 8];
          Bih[j] = bi;
        }
      }

      for (int tt = 0; tt < TC_; ++tt) {
        const int t = c * TC_ + tt;

        // A fragment of h_{t-1} (zeros at global t==0; hcarry at chunk start)
        AFrag ah;
        if (c == 0 && tt == 0) {
          #pragma unroll
          for (int e = 0; e < 16; ++e) ah.v[e] = (_Float16)0.0f;
        } else {
          const _Float16* hs = (tt == 0) ? (hcarry + l * 512) : (hbuf + (tt - 1) * 512);
          const _Float16* hp = hs + lmod * 32 + half * 8;  // row=lmod, K-half per lane group
          ah.h[0] = *(const v8h*)hp;         // K = 8*half .. +7
          ah.h[1] = *(const v8h*)(hp + 16);  // K = 16 + 8*half .. +7
        }

        v8f acc[8];
        if (l == 0) {
          // gates = bias + x_t * w_ih0 (rank-1), then + h W_hh^T via WMMA
          float xv[8];
          #pragma unroll
          for (int v = 0; v < 8; ++v) xv[v] = xbuf[(v + half * 8) * TT_ + t];
          #pragma unroll
          for (int j = 0; j < 8; ++j) {
            v8f a;
            #pragma unroll
            for (int v = 0; v < 8; ++v) a[v] = bias[j] + xv[v] * wx0[j];
            acc[j] = a;
          }
        } else {
          // gates = bias + x_t W_ih^T (WMMA) ...
          AFrag axf;
          const _Float16* hp = hbuf + tt * 512 + lmod * 32 + half * 8;
          axf.h[0] = *(const v8h*)hp;
          axf.h[1] = *(const v8h*)(hp + 16);
          #pragma unroll
          for (int j = 0; j < 8; ++j) {
            v8f a;
            #pragma unroll
            for (int v = 0; v < 8; ++v) a[v] = bias[j];
            acc[j] = WMMA16(axf.v, Bih[j], a);
          }
        }
        // ... + h_{t-1} W_hh^T (WMMA, chained through C)
        #pragma unroll
        for (int j = 0; j < 8; ++j) acc[j] = WMMA16(ah.v, Bhh[j], acc[j]);

        // LSTM cell update on f32 accumulators; gate tiles: i=0,1 f=2,3 g=4,5 o=6,7
        _Float16* hw = hbuf + tt * 512;
        _Float16* hc = hcarry + l * 512;
        #pragma unroll
        for (int v = 0; v < 8; ++v) {
          float ig0 = sigm(acc[0][v]),      ig1 = sigm(acc[1][v]);
          float fg0 = sigm(acc[2][v]),      fg1 = sigm(acc[3][v]);
          float gg0 = tanh_fast(acc[4][v]), gg1 = tanh_fast(acc[5][v]);
          float og0 = sigm(acc[6][v]),      og1 = sigm(acc[7][v]);
          float cc0 = fg0 * c0s[l][v] + ig0 * gg0;
          float cc1 = fg1 * c1s[l][v] + ig1 * gg1;
          c0s[l][v] = cc0; c1s[l][v] = cc1;
          _Float16 h0 = (_Float16)(og0 * tanh_fast(cc0));
          _Float16 h1 = (_Float16)(og1 * tanh_fast(cc1));
          const int so = (v + half * 8) * 32;   // seq row in [16][32] tile
          hw[so + lmod]      = h0;              // H = lmod
          hw[so + 16 + lmod] = h1;              // H = 16 + lmod
          if (tt == TC_ - 1) { hc[so + lmod] = h0; hc[so + 16 + lmod] = h1; }
        }
      }
    }
  }
  __syncthreads();

  // ---- final FC: out[n] = h_last . fc_w + fc_b (h_last = hbuf[TC-1] from layer 2) ----
  if (lane < 16) {
    const _Float16* hl = hbuf + (TC_ - 1) * 512 + lane * 32;
    float a = fcb[0];
    #pragma unroll
    for (int k = 0; k < 32; ++k) a += (float)hl[k] * fcw[k];
    out[seq_base + lane] = a;
  }
}

extern "C" void kernel_launch(void* const* d_in, const int* in_sizes, int n_in,
                              void* d_out, int out_size, void* d_ws, size_t ws_size,
                              hipStream_t stream) {
  const float* x    = (const float*)d_in[0];
  const float* wih0 = (const float*)d_in[1];
  const float* whh0 = (const float*)d_in[2];
  const float* bih0 = (const float*)d_in[3];
  const float* bhh0 = (const float*)d_in[4];
  const float* wih1 = (const float*)d_in[5];
  const float* whh1 = (const float*)d_in[6];
  const float* bih1 = (const float*)d_in[7];
  const float* bhh1 = (const float*)d_in[8];
  const float* wih2 = (const float*)d_in[9];
  const float* whh2 = (const float*)d_in[10];
  const float* bih2 = (const float*)d_in[11];
  const float* bhh2 = (const float*)d_in[12];
  const float* fcw  = (const float*)d_in[13];
  const float* fcb  = (const float*)d_in[14];
  float* out = (float*)d_out;

  const long N    = (long)in_sizes[0] / TT_;   // number of sequences (65536)
  const int  grid = (int)(N / 16);             // 16 sequences per single-wave WG
  const size_t shmem = (size_t)(TC_ + 3) * 16 * H_ * 2 + 16 * TT_ * 4;  // 23552 B

  hipLaunchKernelGGL(lstm3_wmma_kernel, dim3(grid), dim3(32), shmem, stream,
                     x, wih0, whh0, bih0, bhh0,
                     wih1, whh1, bih1, bhh1,
                     wih2, whh2, bih2, bhh2,
                     fcw, fcb, out);
}
